// PairWiseLearning_MVGRL_65532611002854
// MI455X (gfx1250) — compile-verified, matrix operand order checked by reading, not run on hardware
//
#include <hip/hip_runtime.h>
#include <hip/hip_bf16.h>

#define NUM_NODES   262144
#define NUM_GRAPHS  256
#define D_MODEL     256
#define ROWS_PER_WG 128
#define WAVES_PER_WG 8
#define THREADS     256
#define LN2f        0.6931471805599453f
#define NLOG2Ef    (-1.4426950408889634f)

typedef __bf16       v16bf __attribute__((ext_vector_type(16)));
typedef float        v8f   __attribute__((ext_vector_type(8)));
typedef float        v4f   __attribute__((ext_vector_type(4)));
typedef unsigned int v4u   __attribute__((ext_vector_type(4)));

// g_enc staged in LDS as bf16, rows padded 256 -> 264 elems (528 B) so the
// 16 column-lanes of a ds_load_b128 land on distinct bank groups.
#define G_ROW_BF16 264
#define G_BYTES    (NUM_GRAPHS * G_ROW_BF16 * 2)          // 135168 B
#define SMEM_BYTES (G_BYTES + WAVES_PER_WG * sizeof(float))

__device__ __forceinline__ void pack8(v16bf& a, int base, v4f x0, v4f x1) {
    a[base + 0] = (__bf16)x0[0]; a[base + 1] = (__bf16)x0[1];
    a[base + 2] = (__bf16)x0[2]; a[base + 3] = (__bf16)x0[3];
    a[base + 4] = (__bf16)x1[0]; a[base + 5] = (__bf16)x1[1];
    a[base + 6] = (__bf16)x1[2]; a[base + 7] = (__bf16)x1[3];
}

__global__ __launch_bounds__(THREADS)
void mvgrl_jsd_main(const float* __restrict__ l_enc,
                    const float* __restrict__ g_enc,
                    const int*   __restrict__ batch,
                    double*      __restrict__ accum)
{
    extern __shared__ char smem[];
    __bf16* gB  = (__bf16*)smem;
    float*  red = (float*)(smem + G_BYTES);

    const int tid  = threadIdx.x;
    const int lane = tid & 31;
    const int wave = tid >> 5;
    const int m    = lane & 15;   // row (A) / column (B) index within tile
    const int half = lane >> 4;   // selects K sub-blocks per ISA 16-bit layout

    // ---- Stage g_enc (f32, L2-resident) into LDS as bf16, one row/thread ----
    {
        const float* src = g_enc + tid * D_MODEL;
        __bf16*      dst = gB + tid * G_ROW_BF16;
        #pragma unroll
        for (int c = 0; c < D_MODEL; c += 8) {
            v4f x0 = *(const v4f*)(src + c);
            v4f x1 = *(const v4f*)(src + c + 4);
            union { __bf16 h[8]; v4u u; } pk;
            pk.h[0] = (__bf16)x0[0]; pk.h[1] = (__bf16)x0[1];
            pk.h[2] = (__bf16)x0[2]; pk.h[3] = (__bf16)x0[3];
            pk.h[4] = (__bf16)x1[0]; pk.h[5] = (__bf16)x1[1];
            pk.h[6] = (__bf16)x1[2]; pk.h[7] = (__bf16)x1[3];
            *(v4u*)(dst + c) = pk.u;                        // 16-B LDS store
        }
    }
    __syncthreads();

    // ---- Load this wave's 16-row A panel (K = 0..255) as 8 bf16 fragments ----
    // A layout (ISA 7.12.2, 16-bit A 16x32): lanes 0-15 row M=lane hold
    // K = {0..7} u {16..23}; lanes 16-31 same rows hold K = {8..15} u {24..31}.
    const long  rowbase = (long)blockIdx.x * ROWS_PER_WG + wave * 16;
    const float* arow   = l_enc + (rowbase + m) * D_MODEL;

    v16bf A[8];
    #pragma unroll
    for (int kk = 0; kk < 8; ++kk) {
        const int k0 = kk * 32 + half * 8;
        v4f x0 = __builtin_nontemporal_load((const v4f*)(arow + k0));
        v4f x1 = __builtin_nontemporal_load((const v4f*)(arow + k0 + 4));
        v4f y0 = __builtin_nontemporal_load((const v4f*)(arow + k0 + 16));
        v4f y1 = __builtin_nontemporal_load((const v4f*)(arow + k0 + 20));
        v16bf a;
        pack8(a, 0, x0, x1);
        pack8(a, 8, y0, y1);
        A[kk] = a;
    }

    // C/D layout: VGPR i holds (row = rowbase + half*8 + i, col = gt*16 + m)
    int bsel[8];
    #pragma unroll
    for (int i = 0; i < 8; ++i)
        bsel[i] = batch[rowbase + half * 8 + i];

    // ln2 folded into the JSD weights (epilogue works in base-2 throughout)
    const float kpos = LN2f / (float)NUM_NODES;
    const float kneg = LN2f / ((float)NUM_NODES * (float)(NUM_GRAPHS - 1));
    float local = 0.0f;

    #pragma unroll 1
    for (int gt = 0; gt < 16; ++gt) {
        v8f acc = {};
        const __bf16* bcol = gB + (gt * 16 + m) * G_ROW_BF16;
        #pragma unroll
        for (int kk = 0; kk < 8; ++kk) {
            const int k0 = kk * 32 + half * 8;
            union { v4u u[2]; v16bf b; } bu;
            bu.u[0] = *(const v4u*)(bcol + k0);        // K {0..7} sub-block
            bu.u[1] = *(const v4u*)(bcol + k0 + 16);   // K {16..23} sub-block
            acc = __builtin_amdgcn_wmma_f32_16x16x32_bf16(
                      false, A[kk], false, bu.b, (short)0, acc, false, false);
        }
        const int g = gt * 16 + m;
        #pragma unroll
        for (int i = 0; i < 8; ++i) {
            float r = acc[i];
            // Base-2 stable softplus: u = -r*log2e,
            //   softplus(-r)/ln2 = max(u,0) + log2(1 + 2^(-|u|))
            // 2^(-|u|): neg+abs fold into v_exp_f32 src modifiers;
            // v_log_f32 arg is in (1,2] so the bare hw op is exact enough.
            float u  = r * NLOG2Ef;
            float t  = __builtin_amdgcn_exp2f(-fabsf(u));
            float bp = fmaxf(u, 0.0f) + __builtin_amdgcn_logf(1.0f + t) - 1.0f;
            float bn = bp - u;
            // pos: (s - ln2)/ln2 = bp    neg: (s + r - ln2)/ln2 = bn
            local += (bsel[i] == g) ? bp * kpos : bn * kneg;
        }
    }

    // ---- wave shuffle reduction -> LDS -> one f64 atomic per block ----
    #pragma unroll
    for (int off = 16; off > 0; off >>= 1)
        local += __shfl_xor(local, off, 32);
    if (lane == 0) red[wave] = local;
    __syncthreads();
    if (tid == 0) {
        float s = 0.0f;
        #pragma unroll
        for (int w = 0; w < WAVES_PER_WG; ++w) s += red[w];
        atomicAdd(accum, (double)s);   // global_atomic_add_f64
    }
}

__global__ void mvgrl_init(double* accum) {
    if (threadIdx.x == 0 && blockIdx.x == 0) accum[0] = 0.0;
}

__global__ void mvgrl_fin(const double* __restrict__ accum,
                          float* __restrict__ out) {
    if (threadIdx.x == 0 && blockIdx.x == 0) out[0] = (float)accum[0];
}

extern "C" void kernel_launch(void* const* d_in, const int* in_sizes, int n_in,
                              void* d_out, int out_size, void* d_ws, size_t ws_size,
                              hipStream_t stream) {
    const float* l_enc = (const float*)d_in[0];
    const float* g_enc = (const float*)d_in[1];
    const int*   batch = (const int*)d_in[2];
    float*  out   = (float*)d_out;
    double* accum = (double*)d_ws;

    mvgrl_init<<<1, 1, 0, stream>>>(accum);
    mvgrl_jsd_main<<<NUM_NODES / ROWS_PER_WG, THREADS, SMEM_BYTES, stream>>>(
        l_enc, g_enc, batch, accum);
    mvgrl_fin<<<1, 1, 0, stream>>>(accum, out);
}
